// E2EModel_49194555408736
// MI455X (gfx1250) — 16-bit and 8-bit matrix operand order repaired and checked by reading, compiled, not run on hardware
//
#include <hip/hip_runtime.h>
#include <hip/hip_bf16.h>

#define T_STEPS 16384
#define HDIM    256
#define NROWS   1024

typedef __attribute__((ext_vector_type(16))) int            v16i;
typedef __attribute__((ext_vector_type(8)))  float          v8f;
typedef __attribute__((ext_vector_type(16))) __bf16         v16bf;
typedef __attribute__((ext_vector_type(16))) unsigned short v16us;

__device__ inline unsigned short f2bf(float f) {
  unsigned u = __float_as_uint(f);
  unsigned r = ((u >> 16) & 1u) + 0x7fffu;
  return (unsigned short)((u + r) >> 16);
}

__device__ inline unsigned char f32_to_e4m3(float x) {
  unsigned s = (__float_as_uint(x) >> 31) << 7;
  float ax = fabsf(x);
  if (!(ax < 448.f)) return (unsigned char)(s | 0x7e);
  int q = (int)rintf(ax * 512.f);
  if (q <= 7) return (unsigned char)(s | (unsigned)q);
  int e = (int)((__float_as_uint(ax) >> 23) & 0xff) - 127;
  float scale = __uint_as_float((unsigned)(127 + 3 - e) << 23);
  int mq = (int)rintf(ax * scale);
  if (mq >= 16) { mq = 8; ++e; }
  if (e < -6) return (unsigned char)(s | 0x08);
  if (e > 8)  return (unsigned char)(s | 0x7e);
  return (unsigned char)(s | ((unsigned)(e + 7) << 3) | (unsigned)(mq - 8));
}

__device__ inline float sigmoidf_(float x) { return 1.f / (1.f + __expf(-x)); }

__global__ void prep_w_fp8(const float* __restrict__ W_hh,
                           unsigned int* __restrict__ wtile) {
  int fid = blockIdx.x * blockDim.x + threadIdx.x;
  if (fid >= 65536) return;
  int tile = fid >> 9;
  int rem  = fid & 511;
  int lane = rem >> 4;
  int v    = rem & 15;
  int wv = tile >> 3;
  int rt = (tile >> 1) & 3;
  int kt = tile & 1;
  int M    = wv * 64 + rt * 16 + (lane & 15);
  int half = lane >> 4;
  int v0   = v & 7;
  int koff = ((v >> 3) & 1) * 64 + ((v0 >> 2) & 1) * 32 +
             half * 16 + (v0 & 3) * 4;
  int K = kt * 128 + koff;
  const float* row = W_hh + M * HDIM + K;
  unsigned b0 = f32_to_e4m3(row[0]);
  unsigned b1 = f32_to_e4m3(row[1]);
  unsigned b2 = f32_to_e4m3(row[2]);
  unsigned b3 = f32_to_e4m3(row[3]);
  wtile[fid] = b0 | (b1 << 8) | (b2 << 16) | (b3 << 24);
}

__global__ __launch_bounds__(512) void lstm_kernel(
    const float* __restrict__ y_seq, const float* __restrict__ W_ih,
    const float* __restrict__ b_ih,  const float* __restrict__ b_hh,
    const unsigned int* __restrict__ wtile, unsigned short* __restrict__ hs) {
  __shared__ __align__(16) unsigned char hf8[HDIM];
  __shared__ float gates[NROWS];

  const int tid  = threadIdx.x;
  const int lane = tid & 31;
  const int wv   = tid >> 5;
  const int half = lane >> 4;

  v16i A[8];
#pragma unroll
  for (int i = 0; i < 8; ++i) {
    const int* p = (const int*)wtile + (wv * 8 + i) * 512 + lane * 16;
#pragma unroll
    for (int e = 0; e < 16; ++e) A[i][e] = p[e];
  }

  float c = 0.f, h = 0.f;
  float wi[4] = {0.f, 0.f, 0.f, 0.f};
  float bs[4] = {0.f, 0.f, 0.f, 0.f};
  if (tid < HDIM) {
#pragma unroll
    for (int g = 0; g < 4; ++g) {
      wi[g] = W_ih[g * HDIM + tid];
      bs[g] = b_ih[g * HDIM + tid] + b_hh[g * HDIM + tid];
    }
  }
  if (tid < 64) ((int*)hf8)[tid] = 0;
  __syncthreads();

  for (int t = 0; t < T_STEPS; ++t) {
    __builtin_prefetch(y_seq + t + 8, 0, 1);
    v16i B0, B1;
#pragma unroll
    for (int g = 0; g < 4; ++g) {
      const int* q0 = (const int*)(hf8 + g * 32 + half * 16);
      const int* q1 = (const int*)(hf8 + 128 + g * 32 + half * 16);
#pragma unroll
      for (int j = 0; j < 4; ++j) { B0[4 * g + j] = q0[j]; B1[4 * g + j] = q1[j]; }
    }
#pragma unroll
    for (int rt = 0; rt < 4; ++rt) {
      v8f acc = {0.f, 0.f, 0.f, 0.f, 0.f, 0.f, 0.f, 0.f};
      acc = __builtin_amdgcn_wmma_f32_16x16x128_fp8_fp8(A[rt * 2 + 0], B0,
                                                        (short)0, acc, false, false);
      acc = __builtin_amdgcn_wmma_f32_16x16x128_fp8_fp8(A[rt * 2 + 1], B1,
                                                        (short)0, acc, false, false);
      if ((lane & 15) == 0) {
        int mbase = wv * 64 + rt * 16 + half * 8;
#pragma unroll
        for (int p = 0; p < 8; ++p) gates[mbase + p] = acc[p];
      }
    }
    __syncthreads();
    if (tid < HDIM) {
      float yt = y_seq[t];
      float gi = gates[tid]            + wi[0] * yt + bs[0];
      float gf = gates[HDIM + tid]     + wi[1] * yt + bs[1];
      float gg = gates[2 * HDIM + tid] + wi[2] * yt + bs[2];
      float go = gates[3 * HDIM + tid] + wi[3] * yt + bs[3];
      c = sigmoidf_(gf) * c + sigmoidf_(gi) * tanhf(gg);
      h = sigmoidf_(go) * tanhf(c);
      hf8[tid] = f32_to_e4m3(h);
      hs[t * HDIM + tid] = f2bf(h);
    }
    __syncthreads();
  }
}

__global__ __launch_bounds__(128) void gemv_a_kernel(
    const unsigned short* __restrict__ hs, const float* __restrict__ W_fc,
    const float* __restrict__ b_fc, float* __restrict__ a) {
  int lane = threadIdx.x & 31, wv = threadIdx.x >> 5;
  int rtile = blockIdx.x * 4 + wv;
  int t0 = rtile * 16;
  int M = lane & 15, half = lane >> 4;
  v8f acc = {0.f, 0.f, 0.f, 0.f, 0.f, 0.f, 0.f, 0.f};
#pragma unroll
  for (int kt = 0; kt < 8; ++kt) {
    v16us au, bu;
#pragma unroll
    for (int d = 0; d < 8; ++d) {
      int Ka = kt * 32 + (d >> 2) * 16 + (d & 3) * 2 + half * 8;
      const unsigned short* pa = hs + (t0 + M) * HDIM + Ka;
      au[2 * d] = pa[0]; au[2 * d + 1] = pa[1];
      int Kb = kt * 32 + (d >> 2) * 16 + (d & 3) * 2 + half * 8;
      bu[2 * d]     = f2bf(W_fc[Kb]);
      bu[2 * d + 1] = f2bf(W_fc[Kb + 1]);
    }
    acc = __builtin_amdgcn_wmma_f32_16x16x32_bf16(
        false, __builtin_bit_cast(v16bf, au),
        false, __builtin_bit_cast(v16bf, bu), (short)0, acc, false, false);
  }
  if ((lane & 15) == 0) {
    float bb = b_fc[0];
    int base = t0 + half * 8;
#pragma unroll
    for (int p = 0; p < 8; ++p) a[base + p] = acc[p] + bb;
  }
}

__global__ void scan_kernel(const float* __restrict__ y, const float* __restrict__ a,
                            const float* __restrict__ x0, const float* __restrict__ r_theta,
                            const float* __restrict__ W_fc, float* __restrict__ out) {
  if (threadIdx.x != 0 || blockIdx.x != 0) return;
  float theta = log1pf(expf(r_theta[0]));
  float w_x = W_fc[HDIM];
  float denom = 1.f + 5.f + theta + 1e-8f;
  float prev = x0[0];
  float loss = 0.f;
  for (int t = 0; t < T_STEPS; ++t) {
    float at = a[t];
    float yt = y[t];
    float x_ml = at + w_x * prev;
    float x = (5.f * yt + prev + theta * x_ml) / denom;
    float d1 = x - yt, d2 = x - prev;
    loss += 2.5f * d1 * d1 + 0.5f * d2 * d2;
    out[1 + t] = x;
    prev = x;
  }
  out[0] = loss / (float)T_STEPS;
}

extern "C" void kernel_launch(void* const* d_in, const int* in_sizes, int n_in,
                              void* d_out, int out_size, void* d_ws, size_t ws_size,
                              hipStream_t stream) {
  const float* y_seq   = (const float*)d_in[0];
  const float* x0      = (const float*)d_in[1];
  const float* r_theta = (const float*)d_in[2];
  const float* W_ih    = (const float*)d_in[3];
  const float* W_hh    = (const float*)d_in[4];
  const float* b_ih    = (const float*)d_in[5];
  const float* b_hh    = (const float*)d_in[6];
  const float* W_fc    = (const float*)d_in[7];
  const float* b_fc    = (const float*)d_in[8];

  unsigned char* ws   = (unsigned char*)d_ws;
  unsigned int* wtile = (unsigned int*)ws;
  unsigned short* hs  = (unsigned short*)(ws + (256u << 10));
  float* a            = (float*)(ws + (256u << 10) + (8u << 20));
  float* out          = (float*)d_out;

  prep_w_fp8<<<256, 256, 0, stream>>>(W_hh, wtile);
  lstm_kernel<<<1, 512, 0, stream>>>(y_seq, W_ih, b_ih, b_hh, wtile, hs);
  gemv_a_kernel<<<256, 128, 0, stream>>>(hs, W_fc, b_fc, a);
  scan_kernel<<<1, 32, 0, stream>>>(y_seq, a, x0, r_theta, W_fc, out);
}
